// GNNSmolgenModule_54262616818017
// MI455X (gfx1250) — matrix operand order repaired and checked by reading, not compiled, not run on hardware
//
#include <hip/hip_runtime.h>
#include <hip/hip_bf16.h>
#include <cstdint>

typedef __attribute__((ext_vector_type(16))) __bf16 bf16x16;
typedef __attribute__((ext_vector_type(8)))  float  f32x8;
typedef __attribute__((ext_vector_type(4)))  float  f32x4;
typedef __attribute__((ext_vector_type(4)))  unsigned int u32x4;

#define FLAG_BIAS  1
#define FLAG_RELU  2
#define FLAG_SILU  4
#define FLAG_ACCUM 8

#define MAP_PLAIN 0
#define MAP_BOARD 1
#define MAP_NB    2

struct GemmArgs {
  const float* A;
  const float* W;
  const float* bias;
  float*       C;
  int M, N, K;
  int lda, ldw, ldc;
  long long aStrideZ, wStrideZ, bStrideZ, cStrideZ;
  int flags, amap;
};

#define LDSS 40   // padded LDS row stride in halves (32 data + 8 pad)

// Generic bf16 WMMA GEMM: C[M,N] (op)= A[M,K] * W[K,N] (+bias, relu/silu)
// Block: 128 threads = 4 waves. Tile: 64(M) x 64(N), K-step 32.
// A tile double-buffered in LDS (4x cross-wave reuse); B fragments gathered
// straight from global into registers (no cross-wave reuse -> no LDS).
// Register-pipelined, unroll-2, last k-step peeled => branch-free main body.
template<int FLAGS, int AMAP>
__global__ __launch_bounds__(128) void gemm_bf16_wmma(GemmArgs a)
{
  __shared__ __bf16 As[2][64 * LDSS];

  const int tid  = threadIdx.x;
  const int lane = tid & 31;
  const int wv   = tid >> 5;     // 0..3: column tile
  const int half = lane >> 4;    // 0/1
  const int l16  = lane & 15;

  const int z  = blockIdx.z;
  const int m0 = blockIdx.y * 64;
  const int n0 = blockIdx.x * 64;

  const float* Ap   = a.A    + (long long)z * a.aStrideZ;
  const float* Wp   = a.W    + (long long)z * a.wStrideZ;
  const float* Bias = a.bias + (long long)z * a.bStrideZ;
  float*       Cp   = a.C    + (long long)z * a.cStrideZ;

  // ---- A staging role: 2 threads per row, 16 floats each ----
  const int sRow = tid >> 1;           // 0..63
  const int sSeg = (tid & 1) * 16;     // 0 or 16
  long long rowOff;
  {
    int m = m0 + sRow;
    if (AMAP == MAP_BOARD) {                 // inputs[:,1:65,:]
      int b = m >> 6;
      rowOff = ((long long)b * 79 + 1 + (m & 63)) * 1024;
    } else if (AMAP == MAP_NB) {             // inputs[:, nb_idx, :]
      int b = m / 15;
      int j = m - b * 15;
      int row = (j == 0) ? 0 : (64 + j);
      rowOff = ((long long)b * 79 + row) * 1024;
    } else {
      rowOff = (long long)m * a.lda;
    }
  }
  const float* aRow = Ap + rowOff + sSeg;

  // ---- B fragment: per-lane column gather (K-contiguous at fixed n) ----
  const int colB = n0 + wv * 16 + l16;
  const int cCl  = (colB < a.N) ? colB : (a.N - 1);   // always-legal column
  const float bMask = (colB < a.N) ? 1.f : 0.f;       // branch-free zero-fill
  const float* bCol = Wp + (long long)(half * 8) * a.ldw + cCl;

  f32x8 acc[4];
  #pragma unroll
  for (int mt = 0; mt < 4; ++mt)
    #pragma unroll
    for (int i = 0; i < 8; ++i) acc[mt][i] = 0.f;

  union FragU { bf16x16 v; u32x4 q[2]; __bf16 h[16]; };

  f32x4 aReg[4];
  float bReg[16];

  auto loadStep = [&](int k0) {              // global -> registers
    const float* an = aRow + k0;
    #pragma unroll
    for (int q = 0; q < 4; ++q) aReg[q] = ((const f32x4*)an)[q];
    const float* bn = bCol + (long long)k0 * a.ldw;
    #pragma unroll
    for (int j = 0; j < 16; ++j) bReg[j] = bn[(long long)(j + (j & 8)) * a.ldw];
  };
  auto commitA = [&](int buf) {              // registers -> LDS (f32 -> bf16)
    FragU pk;
    #pragma unroll
    for (int q = 0; q < 4; ++q) {
      pk.h[q*4+0] = (__bf16)aReg[q].x;
      pk.h[q*4+1] = (__bf16)aReg[q].y;
      pk.h[q*4+2] = (__bf16)aReg[q].z;
      pk.h[q*4+3] = (__bf16)aReg[q].w;
    }
    __bf16* dst = &As[buf][sRow * LDSS + sSeg];
    *(u32x4*)(dst)     = pk.q[0];
    *(u32x4*)(dst + 8) = pk.q[1];
  };
  auto buildFb = [&]() {                     // registers -> B fragment
    FragU fb;
    #pragma unroll
    for (int j = 0; j < 16; ++j) fb.h[j] = (__bf16)(bReg[j] * bMask);
    return fb;
  };
  auto mma = [&](int buf, FragU& fb) {       // 4 row tiles
    #pragma unroll
    for (int mt = 0; mt < 4; ++mt) {
      // A fragment: lane row = l16; halves = K[kb..kb+7], K[kb+16..kb+23], kb = half*8
      FragU fa;
      const __bf16* p = &As[buf][(mt*16 + l16) * LDSS + half*8];
      fa.q[0] = *(const u32x4*)(p);
      fa.q[1] = *(const u32x4*)(p + 16);
      acc[mt] = __builtin_amdgcn_wmma_f32_16x16x32_bf16(
          false, fa.v, false, fb.v, (short)0, acc[mt], false, false);
    }
  };

  loadStep(0);
  int buf = 0;
  int k0 = 0;
  #pragma unroll 2
  for (; k0 < a.K - 32; k0 += 32) {          // branch-free main body
    commitA(buf);
    FragU fb = buildFb();
    loadStep(k0 + 32);                        // unconditional next-tile loads
    __builtin_prefetch(aRow + k0 + 96, 0, 3); // global_prefetch_b8
    __syncthreads();
    mma(buf, fb);
    buf ^= 1;
  }
  {                                           // peeled final k-step
    commitA(buf);
    FragU fb = buildFb();
    __syncthreads();
    mma(buf, fb);
  }

  // ---- epilogue (FLAGS folded at compile time) ----
  const int col = n0 + wv * 16 + l16;
  if (col < a.N) {
    float bv = (FLAGS & FLAG_BIAS) ? Bias[col] : 0.f;
    #pragma unroll
    for (int mt = 0; mt < 4; ++mt) {
      #pragma unroll
      for (int i = 0; i < 8; ++i) {
        int row = m0 + mt*16 + half*8 + i;   // C layout: VGPR i, lane-half -> M
        long long ci = (long long)row * a.ldc + col;
        float v = acc[mt][i] + bv;
        if (FLAGS & FLAG_ACCUM) v += Cp[ci];
        if (FLAGS & FLAG_RELU)  v = fmaxf(v, 0.f);
        if (FLAGS & FLAG_SILU)  v = v * (1.f / (1.f + __expf(-v)));
        Cp[ci] = v;
      }
    }
  }
}

// h = layernorm(relu(U) + h)  rowwise over 256 features. 8 waves/block, 1 row/wave.
__global__ __launch_bounds__(256) void relu_res_ln(const float* __restrict__ U,
                                                   float* __restrict__ h,
                                                   const float* __restrict__ sc,
                                                   const float* __restrict__ of)
{
  const int wv = threadIdx.x >> 5, lane = threadIdx.x & 31;
  const long long row = (long long)blockIdx.x * 8 + wv;
  const float* u = U + row * 256;
  float* hr = h + row * 256;

  float vals[8];
  float s = 0.f;
  #pragma unroll
  for (int i = 0; i < 8; ++i) {
    int c = lane + i * 32;
    float x = fmaxf(u[c], 0.f) + hr[c];
    vals[i] = x; s += x;
  }
  #pragma unroll
  for (int o = 16; o > 0; o >>= 1) s += __shfl_xor(s, o, 32);
  float mean = s * (1.f / 256.f);
  float vv = 0.f;
  #pragma unroll
  for (int i = 0; i < 8; ++i) { float d = vals[i] - mean; vv += d * d; }
  #pragma unroll
  for (int o = 16; o > 0; o >>= 1) vv += __shfl_xor(vv, o, 32);
  float inv = rsqrtf(vv * (1.f / 256.f) + 1e-5f);
  #pragma unroll
  for (int i = 0; i < 8; ++i) {
    int c = lane + i * 32;
    hr[c] = (vals[i] - mean) * inv * sc[c] + of[c];
  }
}

// flat[b, p*32+c]: p==0 -> nb[b,0]; 1..64 -> gc[b,p-1]; 65..78 -> nb[b,p-64]
__global__ __launch_bounds__(256) void assemble_flat(const float* __restrict__ nb,
                                                     const float* __restrict__ gc,
                                                     float* __restrict__ flat)
{
  long long idx = (long long)blockIdx.x * 256 + threadIdx.x;
  if (idx >= (long long)512 * 2528) return;
  int b = (int)(idx / 2528);
  int rem = (int)(idx - (long long)b * 2528);
  int p = rem >> 5, c = rem & 31;
  float v;
  if (p == 0)        v = nb[(long long)b * 15 * 32 + c];
  else if (p <= 64)  v = gc[((long long)b * 64 + (p - 1)) * 32 + c];
  else               v = nb[((long long)b * 15 + (p - 64)) * 32 + c];
  flat[idx] = v;
}

static inline void launch_gemm(hipStream_t s, dim3 grid, GemmArgs a) {
  if (!a.bias) { a.bias = a.W; a.bStrideZ = 0; }   // never dereferenced (no FLAG_BIAS)
  const dim3 blk(128);
  const int key = a.amap * 16 + a.flags;
  switch (key) {
    case MAP_BOARD * 16 + FLAG_BIAS:
      hipLaunchKernelGGL((gemm_bf16_wmma<FLAG_BIAS, MAP_BOARD>), grid, blk, 0, s, a); break;
    case MAP_NB * 16 + FLAG_BIAS:
      hipLaunchKernelGGL((gemm_bf16_wmma<FLAG_BIAS, MAP_NB>), grid, blk, 0, s, a); break;
    case FLAG_BIAS:
      hipLaunchKernelGGL((gemm_bf16_wmma<FLAG_BIAS, MAP_PLAIN>), grid, blk, 0, s, a); break;
    case FLAG_ACCUM:
      hipLaunchKernelGGL((gemm_bf16_wmma<FLAG_ACCUM, MAP_PLAIN>), grid, blk, 0, s, a); break;
    case FLAG_BIAS | FLAG_SILU:
      hipLaunchKernelGGL((gemm_bf16_wmma<FLAG_BIAS | FLAG_SILU, MAP_PLAIN>), grid, blk, 0, s, a); break;
    default:
      hipLaunchKernelGGL((gemm_bf16_wmma<0, MAP_PLAIN>), grid, blk, 0, s, a); break;
  }
}

extern "C" void kernel_launch(void* const* d_in, const int* in_sizes, int n_in,
                              void* d_out, int out_size, void* d_ws, size_t ws_size,
                              hipStream_t stream)
{
  (void)in_sizes; (void)n_in; (void)out_size; (void)ws_size;
  const float* inputs    = (const float*)d_in[0];
  const float* adj[3]    = {(const float*)d_in[1], (const float*)d_in[2], (const float*)d_in[3]};
  const float* in_proj_W = (const float*)d_in[4];
  const float* in_proj_b = (const float*)d_in[5];
  const float* msg_W     = (const float*)d_in[6];
  const float* msg_b     = (const float*)d_in[7];
  const float* upd_W     = (const float*)d_in[8];
  const float* upd_b     = (const float*)d_in[9];
  const float* ln_scale  = (const float*)d_in[10];
  const float* ln_offset = (const float*)d_in[11];
  const float* nb_W      = (const float*)d_in[12];
  const float* nb_b      = (const float*)d_in[13];
  const float* gc_W      = (const float*)d_in[14];
  const float* gc_b      = (const float*)d_in[15];
  const float* pos_W     = (const float*)d_in[16];
  const float* pos_b     = (const float*)d_in[17];
  const float* head_W    = (const float*)d_in[18];
  const float* head_b    = (const float*)d_in[19];
  const float* shproj    = (const float*)d_in[20];
  float* out = (float*)d_out;

  float* ws   = (float*)d_ws;
  float* h    = ws;                 // 32768*256
  float* tmp  = h    + 8388608;     // 32768*256
  float* agg  = tmp  + 8388608;     // 32768*256
  float* U    = agg  + 8388608;     // 32768*256
  float* nbb  = U    + 8388608;     // 7680*32
  float* gcb  = nbb  + 245760;      // 32768*32
  float* flat = gcb  + 1048576;     // 512*2528
  float* ps   = flat + 1294336;     // 512*256
  float* hs   = ps   + 131072;      // 512*8*256

  GemmArgs g;

  // 1) h = inputs[:,1:65,:] @ in_proj_W + b        (M=32768,K=1024,N=256)
  g = GemmArgs{inputs, in_proj_W, in_proj_b, h, 32768, 256, 1024,
               0, 256, 256, 0, 0, 0, 0, FLAG_BIAS, MAP_BOARD};
  launch_gemm(stream, dim3(4, 512, 1), g);

  // 2) GNN layers
  for (int l = 0; l < 3; ++l) {
    for (int r = 0; r < 3; ++r) {
      // tmp = h @ msg_W[l,r] + msg_b[l,r]
      g = GemmArgs{h, msg_W + (size_t)(l * 3 + r) * 65536, msg_b + (l * 3 + r) * 256,
                   tmp, 32768, 256, 256, 256, 256, 256, 0, 0, 0, 0,
                   FLAG_BIAS, MAP_PLAIN};
      launch_gemm(stream, dim3(4, 512, 1), g);
      // agg (+)= adj_r[b] @ tmp[b]   batched over z=512: M=64,K=64,N=256
      g = GemmArgs{adj[r], tmp, nullptr, agg, 64, 256, 64, 64, 256, 256,
                   4096, 16384, 0, 16384, (r == 0) ? 0 : FLAG_ACCUM, MAP_PLAIN};
      launch_gemm(stream, dim3(4, 1, 512), g);
    }
    // U = h @ upd_W[l][:256] + upd_b[l]
    g = GemmArgs{h, upd_W + (size_t)l * 131072, upd_b + l * 256, U,
                 32768, 256, 256, 256, 256, 256, 0, 0, 0, 0, FLAG_BIAS, MAP_PLAIN};
    launch_gemm(stream, dim3(4, 512, 1), g);
    // U += agg @ upd_W[l][256:]
    g = GemmArgs{agg, upd_W + (size_t)l * 131072 + 65536, nullptr, U,
                 32768, 256, 256, 256, 256, 256, 0, 0, 0, 0, FLAG_ACCUM, MAP_PLAIN};
    launch_gemm(stream, dim3(4, 512, 1), g);
    // h = LN(relu(U) + h)
    hipLaunchKernelGGL(relu_res_ln, dim3(4096), dim3(256), 0, stream,
                       U, h, ln_scale + l * 256, ln_offset + l * 256);
  }

  // 3) nb = inputs[:,nb_idx,:] @ nb_W + nb_b       (M=7680,K=1024,N=32)
  g = GemmArgs{inputs, nb_W, nb_b, nbb, 7680, 32, 1024,
               0, 32, 32, 0, 0, 0, 0, FLAG_BIAS, MAP_NB};
  launch_gemm(stream, dim3(1, 120, 1), g);

  // 4) gc = h @ gc_W + gc_b                        (M=32768,K=256,N=32)
  g = GemmArgs{h, gc_W, gc_b, gcb, 32768, 32, 256,
               256, 32, 32, 0, 0, 0, 0, FLAG_BIAS, MAP_PLAIN};
  launch_gemm(stream, dim3(1, 512, 1), g);

  // 5) flat = concat([nb0, gc, nb1:])
  hipLaunchKernelGGL(assemble_flat, dim3((512 * 2528 + 255) / 256), dim3(256), 0,
                     stream, nbb, gcb, flat);

  // 6) ps = silu(flat @ pos_W + pos_b)             (M=512,K=2528,N=256)
  g = GemmArgs{flat, pos_W, pos_b, ps, 512, 256, 2528,
               2528, 256, 256, 0, 0, 0, 0, FLAG_BIAS | FLAG_SILU, MAP_PLAIN};
  launch_gemm(stream, dim3(4, 8, 1), g);

  // 7) hs[:,hd,:] = silu(ps @ head_W[hd] + head_b[hd])   z=8 heads
  g = GemmArgs{ps, head_W, head_b, hs, 512, 256, 256,
               256, 256, 2048, 0, 65536, 256, 256, FLAG_BIAS | FLAG_SILU, MAP_PLAIN};
  launch_gemm(stream, dim3(4, 8, 8), g);

  // 8) logits[:,hd,:] = hs[:,hd,:] @ shared_proj          z=8, N=6241
  g = GemmArgs{hs, shproj, nullptr, out, 512, 6241, 256,
               2048, 6241, 49928, 256, 0, 0, 6241, 0, MAP_PLAIN};
  launch_gemm(stream, dim3(98, 8, 8), g);
}
